// MultiDANN_2448131358817
// MI455X (gfx1250) — matrix operand order repaired and checked by reading, not compiled, not run on hardware
//
#include <hip/hip_runtime.h>
#include <hip/hip_bf16.h>

typedef __attribute__((ext_vector_type(16))) _Float16 v16h;
typedef __attribute__((ext_vector_type(8)))  _Float16 v8h;
typedef __attribute__((ext_vector_type(8)))  float    v8f;

#define NN      50000
#define EE      800000
#define TOT_E   (EE + NN)      // edges + self-loops
#define HH      4
#define CC      32
#define HC      128            // H*C
#define GG      64
#define HID     128
#define NCLS    8
#define NEXP    3
#define NEG_SLOPE 0.2f

// ---------- helpers ----------
__device__ __forceinline__ unsigned fkey(float f) {
    unsigned u = __float_as_uint(f);
    return (u & 0x80000000u) ? ~u : (u | 0x80000000u);
}
__device__ __forceinline__ float fdecode(unsigned k) {
    unsigned u = (k & 0x80000000u) ? (k & 0x7FFFFFFFu) : ~k;
    return __uint_as_float(u);
}
__device__ __forceinline__ void edge_sd(const int* ei, int e, int& s, int& d) {
    if (e < EE) { s = ei[e]; d = ei[EE + e]; } else { s = d = e - EE; }
}

__global__ void fill_u32_kernel(unsigned* __restrict__ p, unsigned v, int n) {
    int t = blockIdx.x * blockDim.x + threadIdx.x;
    if (t < n) p[t] = v;
}
__global__ void cvt_f16_kernel(const float* __restrict__ src, _Float16* __restrict__ dst, int n) {
    int t = blockIdx.x * blockDim.x + threadIdx.x;
    if (t < n) dst[t] = (_Float16)src[t];
}
// W [K,128] f32 -> Wt [128,K] f16 (coalesced writes)
__global__ void cvt_w_t_kernel(const float* __restrict__ W, _Float16* __restrict__ Wt, int K) {
    int t = blockIdx.x * blockDim.x + threadIdx.x;
    if (t >= K * 128) return;
    int n = t / K, k = t - n * K;
    Wt[t] = (_Float16)W[k * 128 + n];
}

// ---------- WMMA GEMM: C[M,128] = A[M,K](f16,row-major) x Wt[128,K](f16,B transposed) ----------
// LDS-free: every fragment piece is a contiguous, 16B-aligned global_load_b128.
template<int K>
__global__ __launch_bounds__(256) void gemm_wmma_kernel(
    const _Float16* __restrict__ A, const _Float16* __restrict__ Bt, float* __restrict__ C)
{
    const int tid  = threadIdx.x;
    const int lane = tid & 31;
    const int wave = tid >> 5;            // 0..7 -> N strip
    const int col0 = wave * 16;
    const int row0 = blockIdx.x * 16;
    const int l15  = lane & 15;
    const int hi   = lane >> 4;           // 0 or 1

    // A frag (ISA 7.12.2): lane(m=l15, kb=hi*8): halves [kb..kb+7] and [kb+16..kb+23]
    const _Float16* arow = A  + (size_t)(row0 + l15) * K + hi * 8;
    // B frag: lane(n=l15, kb=hi*8): halves [kb..kb+7] and [kb+16..kb+23] along Wt row
    const _Float16* bcol = Bt + (size_t)(col0 + l15) * K + hi * 8;

    v8f acc = {};
    #pragma unroll
    for (int ks = 0; ks < K; ks += 32) {
        union { v8h h8[2]; v16h h16; } ua, ub;
        ua.h8[0] = *(const v8h*)(arow + ks);        // global_load_b128
        ua.h8[1] = *(const v8h*)(arow + ks + 16);
        ub.h8[0] = *(const v8h*)(bcol + ks);
        ub.h8[1] = *(const v8h*)(bcol + ks + 16);
        acc = __builtin_amdgcn_wmma_f32_16x16x32_f16(false, ua.h16, false, ub.h16,
                                                     (short)0, acc, false, false);
    }
    // D: VGPR v -> row v + hi*8, col l15
    const int m0 = hi * 8;
    #pragma unroll
    for (int v = 0; v < 8; ++v)
        C[(size_t)(row0 + m0 + v) * 128 + col0 + l15] = acc[v];
}

// ---------- attention scores ----------
__global__ void att_scores_kernel(const float* __restrict__ h,
                                  const float* __restrict__ att_s,
                                  const float* __restrict__ att_d,
                                  float* __restrict__ a_src, float* __restrict__ a_dst)
{
    int t = blockIdx.x * blockDim.x + threadIdx.x;
    if (t >= NN * HH) return;
    int n = t >> 2, hh = t & 3;
    const float* hp = h + (size_t)n * HC + hh * CC;
    float ss = 0.f, dd = 0.f;
    #pragma unroll 8
    for (int c = 0; c < CC; ++c) {
        float v = hp[c];
        ss += v * att_s[hh * CC + c];
        dd += v * att_d[hh * CC + c];
    }
    a_src[t] = ss; a_dst[t] = dd;
}

// ---------- edge passes ----------
__global__ void edge_max_kernel(const int* __restrict__ ei,
                                const float* __restrict__ a_src, const float* __restrict__ a_dst,
                                unsigned* __restrict__ emax_u)
{
    int t = blockIdx.x * blockDim.x + threadIdx.x;
    if (t >= TOT_E * 4) return;
    int e = t >> 2, hh = t & 3, s, d;
    edge_sd(ei, e, s, d);
    float v = a_src[s * 4 + hh] + a_dst[d * 4 + hh];
    v = (v > 0.f) ? v : NEG_SLOPE * v;
    atomicMax(emax_u + d * 4 + hh, fkey(v));
}

__global__ void edge_sum_kernel(const int* __restrict__ ei,
                                const float* __restrict__ a_src, const float* __restrict__ a_dst,
                                const unsigned* __restrict__ emax_u, float* __restrict__ denom)
{
    int t = blockIdx.x * blockDim.x + threadIdx.x;
    if (t >= TOT_E * 4) return;
    int e = t >> 2, hh = t & 3, s, d;
    edge_sd(ei, e, s, d);
    float v = a_src[s * 4 + hh] + a_dst[d * 4 + hh];
    v = (v > 0.f) ? v : NEG_SLOPE * v;
    float ee = __expf(v - fdecode(emax_u[d * 4 + hh]));
    atomicAdd(denom + d * 4 + hh, ee);
}

// one float4 of h[src] per thread, 4 atomic f32 adds into agg[dst]
__global__ void edge_scatter_kernel(const int* __restrict__ ei,
                                    const float* __restrict__ a_src, const float* __restrict__ a_dst,
                                    const unsigned* __restrict__ emax_u, const float* __restrict__ denom,
                                    const float* __restrict__ h, float* __restrict__ agg)
{
    long long t = (long long)blockIdx.x * blockDim.x + threadIdx.x;
    if (t >= (long long)TOT_E * 32) return;
    int e  = (int)(t >> 5);
    int c4 = (int)(t & 31) * 4;        // channel base 0,4,...,124
    int hh = c4 >> 5;
    int s, d;
    edge_sd(ei, e, s, d);
    float v = a_src[s * 4 + hh] + a_dst[d * 4 + hh];
    v = (v > 0.f) ? v : NEG_SLOPE * v;
    float ee    = __expf(v - fdecode(emax_u[d * 4 + hh]));
    float alpha = ee / (denom[d * 4 + hh] + 1e-16f);
    const float4 hv = *(const float4*)(h + (size_t)s * HC + c4);
    float* ap = agg + (size_t)d * HC + c4;
    atomicAdd(ap + 0, hv.x * alpha);
    atomicAdd(ap + 1, hv.y * alpha);
    atomicAdd(ap + 2, hv.z * alpha);
    atomicAdd(ap + 3, hv.w * alpha);
}

// ---------- bias / activation ----------
__global__ void bias_relu_f16_kernel(const float* __restrict__ agg, const float* __restrict__ b,
                                     _Float16* __restrict__ out16)
{
    int t = blockIdx.x * blockDim.x + threadIdx.x;
    if (t >= NN * HC) return;
    float v = agg[t] + b[t & 127];
    out16[t] = (_Float16)(v > 0.f ? v : 0.f);
}
__global__ void bias_add_kernel(float* __restrict__ agg, const float* __restrict__ b)
{
    int t = blockIdx.x * blockDim.x + threadIdx.x;
    if (t >= NN * HC) return;
    agg[t] += b[t & 127];
}

// ---------- pooling ----------
__global__ void count_kernel(const int* __restrict__ batch, float* __restrict__ cnt)
{
    int n = blockIdx.x * blockDim.x + threadIdx.x;
    if (n < NN) atomicAdd(cnt + batch[n], 1.0f);
}
__global__ void pool_scatter_kernel(const float* __restrict__ out2, const int* __restrict__ batch,
                                    float* __restrict__ msum, unsigned* __restrict__ mmax)
{
    int t = blockIdx.x * blockDim.x + threadIdx.x;
    if (t >= NN * HC) return;
    int n = t >> 7, c = t & 127;
    int g = batch[n];
    float v = out2[t];
    atomicAdd(msum + g * HC + c, v);
    atomicMax(mmax + g * HC + c, fkey(v));
}
__global__ void feat_norm_kernel(const float* __restrict__ msum, const unsigned* __restrict__ mmax,
                                 const float* __restrict__ cnt, float* __restrict__ feat)
{
    __shared__ float red[256];
    int g = blockIdx.x, t = threadIdx.x;     // 256 threads = 256 features
    float v;
    if (t < HC) v = msum[g * HC + t] / fmaxf(cnt[g], 1.0f);
    else        v = fdecode(mmax[g * HC + (t - HC)]);
    red[t] = v * v;
    __syncthreads();
    for (int s = 128; s > 0; s >>= 1) { if (t < s) red[t] += red[t + s]; __syncthreads(); }
    float nrm = fmaxf(sqrtf(red[0]), 1e-12f);
    feat[g * 256 + t] = v / nrm;
}

// ---------- MLP heads ----------
__global__ void head_hidden_kernel(const float* __restrict__ feat, const float* __restrict__ W1,
                                   const float* __restrict__ b1, float* __restrict__ hid)
{
    int t = blockIdx.x * blockDim.x + threadIdx.x;
    if (t >= GG * HID) return;
    int g = t >> 7, k = t & 127;
    const float* f = feat + g * 256;
    float s = b1[k];
    #pragma unroll 8
    for (int j = 0; j < 256; ++j) s += f[j] * W1[j * HID + k];
    hid[t] = s > 0.f ? s : 0.f;
}
__global__ void head_out_kernel(const float* __restrict__ hid, const float* __restrict__ Wo,
                                const float* __restrict__ bo, float* __restrict__ out, int n_out)
{
    int t = blockIdx.x * blockDim.x + threadIdx.x;
    if (t >= GG * n_out) return;
    int g = t / n_out, o = t % n_out;
    float s = bo[o];
    #pragma unroll 8
    for (int k = 0; k < HID; ++k) s += hid[g * HID + k] * Wo[k * n_out + o];
    out[g * n_out + o] = s;
}

// ---------- host ----------
static inline size_t alignup(size_t x) { return (x + 255) & ~(size_t)255; }
static inline int grid1(long long n, int b) { return (int)((n + b - 1) / b); }

extern "C" void kernel_launch(void* const* d_in, const int* in_sizes, int n_in,
                              void* d_out, int out_size, void* d_ws, size_t ws_size,
                              hipStream_t stream) {
    (void)in_sizes; (void)n_in; (void)out_size; (void)ws_size;
    const float* x      = (const float*)d_in[0];
    const int*   ei     = (const int*)d_in[1];
    const int*   batch  = (const int*)d_in[2];
    const float* eW1    = (const float*)d_in[4];
    const float* eas1   = (const float*)d_in[5];
    const float* ead1   = (const float*)d_in[6];
    const float* eb1    = (const float*)d_in[7];
    const float* eW2    = (const float*)d_in[8];
    const float* eas2   = (const float*)d_in[9];
    const float* ead2   = (const float*)d_in[10];
    const float* eb2    = (const float*)d_in[11];
    const float* cW1    = (const float*)d_in[12];
    const float* cb1    = (const float*)d_in[13];
    const float* cWo    = (const float*)d_in[14];
    const float* cbo    = (const float*)d_in[15];
    const float* dW1    = (const float*)d_in[16];
    const float* db1    = (const float*)d_in[17];
    const float* dWo    = (const float*)d_in[18];
    const float* dbo    = (const float*)d_in[19];
    float* out = (float*)d_out;

    // workspace bump allocator
    char* ws = (char*)d_ws; size_t off = 0;
    auto alloc = [&](size_t bytes) { char* p = ws + off; off += alignup(bytes); return p; };
    float*     h      = (float*)    alloc((size_t)NN * HC * 4);   // GEMM output
    float*     agg    = (float*)    alloc((size_t)NN * HC * 4);   // aggregation / out
    _Float16*  in16   = (_Float16*) alloc((size_t)NN * HC * 2);   // GEMM input (f16)
    _Float16*  Wt1    = (_Float16*) alloc(128 * 64 * 2);          // enc_W1^T f16 [128,64]
    _Float16*  Wt2    = (_Float16*) alloc(128 * 128 * 2);         // enc_W2^T f16 [128,128]
    float*     a_src  = (float*)    alloc((size_t)NN * 4 * 4);
    float*     a_dst  = (float*)    alloc((size_t)NN * 4 * 4);
    unsigned*  emax_u = (unsigned*) alloc((size_t)NN * 4 * 4);
    float*     denom  = (float*)    alloc((size_t)NN * 4 * 4);
    float*     msum   = (float*)    alloc(GG * HC * 4);
    unsigned*  mmax   = (unsigned*) alloc(GG * HC * 4);
    float*     cnt    = (float*)    alloc(GG * 4);
    float*     feat   = (float*)    alloc(GG * 256 * 4);
    float*     hid    = (float*)    alloc(GG * HID * 4);

    const int B = 256;
    const unsigned NEG_INF_KEY = 0x007FFFFFu;   // fkey(-inf)

    // convert input to f16; weights to transposed f16
    cvt_f16_kernel<<<grid1((long long)NN * 64, B), B, 0, stream>>>(x, in16, NN * 64);
    cvt_w_t_kernel<<<grid1(64 * 128, B), B, 0, stream>>>(eW1, Wt1, 64);
    cvt_w_t_kernel<<<grid1(128 * 128, B), B, 0, stream>>>(eW2, Wt2, 128);

    for (int layer = 0; layer < 2; ++layer) {
        // h = in16 @ W (WMMA, LDS-free)
        if (layer == 0)
            gemm_wmma_kernel<64> <<<NN / 16, 256, 0, stream>>>(in16, Wt1, h);
        else
            gemm_wmma_kernel<128><<<NN / 16, 256, 0, stream>>>(in16, Wt2, h);

        const float* as = layer ? eas2 : eas1;
        const float* ad = layer ? ead2 : ead1;
        att_scores_kernel<<<grid1((long long)NN * 4, B), B, 0, stream>>>(h, as, ad, a_src, a_dst);

        fill_u32_kernel<<<grid1(NN * 4, B), B, 0, stream>>>(emax_u, NEG_INF_KEY, NN * 4);
        fill_u32_kernel<<<grid1(NN * 4, B), B, 0, stream>>>((unsigned*)denom, 0u, NN * 4);
        fill_u32_kernel<<<grid1((long long)NN * HC, B), B, 0, stream>>>((unsigned*)agg, 0u, NN * HC);

        edge_max_kernel<<<grid1((long long)TOT_E * 4, B), B, 0, stream>>>(ei, a_src, a_dst, emax_u);
        edge_sum_kernel<<<grid1((long long)TOT_E * 4, B), B, 0, stream>>>(ei, a_src, a_dst, emax_u, denom);
        edge_scatter_kernel<<<grid1((long long)TOT_E * 32, B), B, 0, stream>>>(
            ei, a_src, a_dst, emax_u, denom, h, agg);

        if (layer == 0) {
            // out1 = relu(agg + b1) -> f16 GEMM input for layer 2
            bias_relu_f16_kernel<<<grid1((long long)NN * HC, B), B, 0, stream>>>(agg, eb1, in16);
        } else {
            // out2 = agg + b2 (in place, f32 for pooling)
            bias_add_kernel<<<grid1((long long)NN * HC, B), B, 0, stream>>>(agg, eb2);
        }
    }

    // pooling
    fill_u32_kernel<<<grid1(GG * HC, B), B, 0, stream>>>((unsigned*)msum, 0u, GG * HC);
    fill_u32_kernel<<<grid1(GG * HC, B), B, 0, stream>>>(mmax, NEG_INF_KEY, GG * HC);
    fill_u32_kernel<<<1, B, 0, stream>>>((unsigned*)cnt, 0u, GG);
    count_kernel<<<grid1(NN, B), B, 0, stream>>>(batch, cnt);
    pool_scatter_kernel<<<grid1((long long)NN * HC, B), B, 0, stream>>>(agg, batch, msum, mmax);
    feat_norm_kernel<<<GG, 256, 0, stream>>>(msum, mmax, cnt, feat);

    // heads: label_pred [64,8] then batch_pred [64,3], concatenated flat
    head_hidden_kernel<<<grid1(GG * HID, B), B, 0, stream>>>(feat, cW1, cb1, hid);
    head_out_kernel<<<grid1(GG * NCLS, B), B, 0, stream>>>(hid, cWo, cbo, out, NCLS);
    head_hidden_kernel<<<grid1(GG * HID, B), B, 0, stream>>>(feat, dW1, db1, hid);
    head_out_kernel<<<grid1(GG * NEXP, B), B, 0, stream>>>(hid, dWo, dbo, out + GG * NCLS, NEXP);
}
